// MOELayer_1116691497149
// MI455X (gfx1250) — matrix operand order repaired and checked by reading, not compiled, not run on hardware
//
#include <hip/hip_runtime.h>
#include <stdint.h>

// Problem constants (match reference)
#define D_DIM  2048
#define H_DIM  5632
#define E_NUM  8
#define NT     4096            // B*S tokens
#define CAP    NT              // per-expert token capacity
#define BM     128             // token tile
#define BN     64              // output-column tile
#define BK     32              // K step (bf16 WMMA K)
#define HDN_ROWS (2*NT + BM)   // packed assignments + tail slack (8320)

typedef __bf16 bf16;
typedef __attribute__((ext_vector_type(16))) __bf16       v16bf;
typedef __attribute__((ext_vector_type(2)))  __bf16       v2bf;
typedef __attribute__((ext_vector_type(8)))  float        v8f;
typedef __attribute__((ext_vector_type(4)))  float        v4f;
typedef __attribute__((ext_vector_type(2)))  float        v2f;
typedef __attribute__((ext_vector_type(4)))  unsigned int v4u;
typedef __attribute__((ext_vector_type(4)))  unsigned int u32x4;
typedef __attribute__((ext_vector_type(8)))  int          i32x8;
typedef __attribute__((ext_vector_type(4)))  int          i32x4;

#if defined(__gfx1250__) && __has_builtin(__builtin_amdgcn_tensor_load_to_lds)
#define MOE_TDM 1
#else
#define MOE_TDM 0
#endif

union Frag { v16bf v; v4u q[2]; };

#define WMMA_BF16(a, b, c) \
    __builtin_amdgcn_wmma_f32_16x16x32_bf16(false, (a), false, (b), (short)0, (c), false, false)

// Packed f32x2 -> bf16x2 (single v_cvt_pk_bf16_f32)
__device__ __forceinline__ uint32_t pk2bf(float a, float b) {
    union { v2bf h; uint32_t u; } r;
    v2f t; t.x = a; t.y = b;
    r.h = __builtin_convertvector(t, v2bf);
    return r.u;
}

// A fragment (16x32 bf16) from row-major LDS tile, row stride 20 u32 (80B, bank-spread, 16B aligned)
__device__ __forceinline__ v16bf ldsA(const uint32_t* base, int mbase, int lane) {
    Frag f;
    const v4u* p = (const v4u*)(base + (mbase + (lane & 15)) * 20 + ((lane >> 4) << 2));
    f.q[0] = p[0];   // K 0..7   (lanes 0-15) / K 8..15  (lanes 16-31)
    f.q[1] = p[2];   // K 16..23 (lanes 0-15) / K 24..31 (lanes 16-31)
    return f.v;
}

// B fragment (32x16 bf16) from column-major LDS tile, col stride 20 u32
__device__ __forceinline__ v16bf ldsB(const uint32_t* base, int nbase, int lane) {
    Frag f;
    const v4u* p = (const v4u*)(base + (nbase + (lane & 15)) * 20 + ((lane >> 4) << 2));
    f.q[0] = p[0];   // K 0..7   / K 8..15
    f.q[1] = p[2];   // K 16..23 / K 24..31
    return f.v;
}

// ---------------------------------------------------------------------------
// Kernel 1: zero output + routing counters (required every launch: atomics)
// ---------------------------------------------------------------------------
__global__ void moe_zero(float* __restrict__ out, int* __restrict__ counts) {
    size_t i = (size_t)blockIdx.x * blockDim.x + threadIdx.x;
    const size_t n4 = (size_t)NT * D_DIM / 4;
    v4f z = {0.f, 0.f, 0.f, 0.f};
    for (size_t j = i; j < n4; j += (size_t)gridDim.x * blockDim.x)
        ((v4f*)out)[j] = z;
    if (i < E_NUM) counts[i] = 0;
}

// ---------------------------------------------------------------------------
// Kernel 2: gate — one wave32 per token: logits, top-2, softmax, scatter
// ---------------------------------------------------------------------------
__global__ __launch_bounds__(256) void moe_gate(
    const float* __restrict__ x, const float* __restrict__ wg,
    const float* __restrict__ bg, int* __restrict__ counts,
    int* __restrict__ token_ids, float* __restrict__ token_wts)
{
    const int token = blockIdx.x * 8 + (threadIdx.x >> 5);
    const int lane  = threadIdx.x & 31;
    const float* xr = x + (size_t)token * D_DIM;

    float acc[E_NUM];
#pragma unroll
    for (int q = 0; q < E_NUM; ++q) acc[q] = 0.f;

    for (int d = lane; d < D_DIM; d += 32) {
        float xv = xr[d];
        v4f wlo = *(const v4f*)(wg + (size_t)d * E_NUM);
        v4f whi = *(const v4f*)(wg + (size_t)d * E_NUM + 4);
        acc[0] += xv * wlo.x; acc[1] += xv * wlo.y;
        acc[2] += xv * wlo.z; acc[3] += xv * wlo.w;
        acc[4] += xv * whi.x; acc[5] += xv * whi.y;
        acc[6] += xv * whi.z; acc[7] += xv * whi.w;
    }
#pragma unroll
    for (int m = 16; m > 0; m >>= 1)
#pragma unroll
        for (int q = 0; q < E_NUM; ++q)
            acc[q] += __shfl_xor(acc[q], m, 32);

    if (lane == 0) {
        float lg[E_NUM];
#pragma unroll
        for (int q = 0; q < E_NUM; ++q) lg[q] = acc[q] + bg[q];
        int i0 = 0; float v0 = lg[0];
#pragma unroll
        for (int q = 1; q < E_NUM; ++q) if (lg[q] > v0) { v0 = lg[q]; i0 = q; }
        int i1 = (i0 == 0) ? 1 : 0; float v1 = -3.4e38f;
#pragma unroll
        for (int q = 0; q < E_NUM; ++q) if (q != i0 && lg[q] > v1) { v1 = lg[q]; i1 = q; }
        float e1 = __expf(v1 - v0);
        float w0 = 1.0f / (1.0f + e1);
        float w1 = e1 * w0;
        int p0 = atomicAdd(&counts[i0], 1);
        token_ids[i0 * CAP + p0] = token; token_wts[i0 * CAP + p0] = w0;
        int p1 = atomicAdd(&counts[i1], 1);
        token_ids[i1 * CAP + p1] = token; token_wts[i1 * CAP + p1] = w1;
    }
}

// ---------------------------------------------------------------------------
// Kernel 3: exclusive prefix sum of counts -> packed hdn row offsets
// ---------------------------------------------------------------------------
__global__ void moe_offsets(const int* __restrict__ counts, int* __restrict__ offsets) {
    if (threadIdx.x == 0 && blockIdx.x == 0) {
        int r = 0;
#pragma unroll
        for (int q = 0; q < E_NUM; ++q) { offsets[q] = r; r += counts[q]; }
    }
}

// ---------------------------------------------------------------------------
// Kernel 4: hdn = silu(Xg @ W1) * (Xg @ W3), gathered per expert, bf16 out
// grid: (H/BN, CAP/BM, E), block: 256 (8 waves, 4 row-groups x 2 col-groups)
// ---------------------------------------------------------------------------
__global__ __launch_bounds__(256) void moe_ffn1(
    const float* __restrict__ x,  const float* __restrict__ w1,
    const float* __restrict__ w3, const int* __restrict__ counts,
    const int* __restrict__ offsets, const int* __restrict__ token_ids,
    bf16* __restrict__ hdn)
{
    const int e    = blockIdx.z;
    const int cnt  = counts[e];
    const int tile = blockIdx.y;
    if (tile * BM >= cnt) return;
    const int h0  = blockIdx.x * BN;
    const int off = offsets[e];

    const float* W1 = w1 + (size_t)e * D_DIM * H_DIM;
    const float* W3 = w3 + (size_t)e * D_DIM * H_DIM;

    __shared__ __align__(16) uint32_t As [BM * 20];
    __shared__ __align__(16) uint32_t Bs1[BN * 20];
    __shared__ __align__(16) uint32_t Bs3[BN * 20];

    const int tid  = threadIdx.x;
    const int lane = tid & 31;
    const int wave = tid >> 5;
    const int wr   = wave & 3;   // row group: rows wr*32 .. +31
    const int wc   = wave >> 2;  // col group: cols wc*32 .. +31

    // A loader: 2 threads per row (16 f32 each)
    const int arow  = tid >> 1;
    const int ahalf = tid & 1;
    const int grow  = tile * BM + arow;
    const float* xrow = nullptr;
    if (grow < cnt) {
        int tok = token_ids[e * CAP + grow];
        xrow = x + (size_t)tok * D_DIM + ahalf * 16;
    }
    // B loader: column bc, k-subgroup bg (8 k values)
    const int bc = tid & 63;
    const int bg = tid >> 6;

    v8f z8 = {0.f,0.f,0.f,0.f,0.f,0.f,0.f,0.f};
    v8f c1[2][2], c3[2][2];
#pragma unroll
    for (int i = 0; i < 2; ++i)
#pragma unroll
        for (int j = 0; j < 2; ++j) { c1[i][j] = z8; c3[i][j] = z8; }

    for (int k0 = 0; k0 < D_DIM; k0 += BK) {
        // ---- stage A (gathered token rows, f32 -> packed bf16) ----
        v4u apk0, apk1;
        if (xrow) {
            v4f f0 = *(const v4f*)(xrow + k0 + 0);
            v4f f1 = *(const v4f*)(xrow + k0 + 4);
            v4f f2 = *(const v4f*)(xrow + k0 + 8);
            v4f f3 = *(const v4f*)(xrow + k0 + 12);
            apk0.x = pk2bf(f0.x,f0.y); apk0.y = pk2bf(f0.z,f0.w);
            apk0.z = pk2bf(f1.x,f1.y); apk0.w = pk2bf(f1.z,f1.w);
            apk1.x = pk2bf(f2.x,f2.y); apk1.y = pk2bf(f2.z,f2.w);
            apk1.z = pk2bf(f3.x,f3.y); apk1.w = pk2bf(f3.z,f3.w);
        } else {
            apk0.x = apk0.y = apk0.z = apk0.w = 0u;
            apk1 = apk0;
        }
        // ---- stage B panels (coalesced across lanes at fixed k) ----
        float b1v[8], b3v[8];
        const size_t gbase = (size_t)(k0 + bg * 8) * H_DIM + h0 + bc;
#pragma unroll
        for (int i = 0; i < 8; ++i) {
            b1v[i] = W1[gbase + (size_t)i * H_DIM];
            b3v[i] = W3[gbase + (size_t)i * H_DIM];
        }
        if (k0 + BK < D_DIM) {
            __builtin_prefetch(&W1[gbase + (size_t)BK * H_DIM], 0, 1);
            __builtin_prefetch(&W3[gbase + (size_t)BK * H_DIM], 0, 1);
        }
        // previous iteration's consumers are done (end-of-loop barrier)
        *(v4u*)(As + arow * 20 + ahalf * 8)     = apk0;
        *(v4u*)(As + arow * 20 + ahalf * 8 + 4) = apk1;
        v4u bp;
        bp.x = pk2bf(b1v[0],b1v[1]); bp.y = pk2bf(b1v[2],b1v[3]);
        bp.z = pk2bf(b1v[4],b1v[5]); bp.w = pk2bf(b1v[6],b1v[7]);
        *(v4u*)(Bs1 + bc * 20 + bg * 4) = bp;
        bp.x = pk2bf(b3v[0],b3v[1]); bp.y = pk2bf(b3v[2],b3v[3]);
        bp.z = pk2bf(b3v[4],b3v[5]); bp.w = pk2bf(b3v[6],b3v[7]);
        *(v4u*)(Bs3 + bc * 20 + bg * 4) = bp;
        __syncthreads();

        v16bf a0 = ldsA(As, wr * 32 + 0,  lane);
        v16bf a1 = ldsA(As, wr * 32 + 16, lane);
#pragma unroll
        for (int nt = 0; nt < 2; ++nt) {
            v16bf bf1 = ldsB(Bs1, wc * 32 + nt * 16, lane);
            c1[0][nt] = WMMA_BF16(a0, bf1, c1[0][nt]);
            c1[1][nt] = WMMA_BF16(a1, bf1, c1[1][nt]);
            v16bf bf3 = ldsB(Bs3, wc * 32 + nt * 16, lane);
            c3[0][nt] = WMMA_BF16(a0, bf3, c3[0][nt]);
            c3[1][nt] = WMMA_BF16(a1, bf3, c3[1][nt]);
        }
        __syncthreads();
    }

    // ---- epilogue: h = silu(c1) * c3, store bf16 ----
    const int colb = h0 + wc * 32 + (lane & 15);
    const int rsub = (lane >> 4) * 8;
#pragma unroll
    for (int mt = 0; mt < 2; ++mt) {
        const int lrow = wr * 32 + mt * 16 + rsub;
#pragma unroll
        for (int nt = 0; nt < 2; ++nt) {
            v8f u = c1[mt][nt];
            v8f g = c3[mt][nt];
            const int col = colb + nt * 16;
#pragma unroll
            for (int i = 0; i < 8; ++i) {
                int gr = tile * BM + lrow + i;
                if (gr < cnt) {
                    float uv = u[i];
                    float hv = (uv / (1.0f + __expf(-uv))) * g[i];
                    hdn[(size_t)(off + gr) * H_DIM + col] = (bf16)hv;
                }
            }
        }
    }
}

// ---------------------------------------------------------------------------
// Kernel 5: out += weight * (hdn @ W2), scatter with f32 atomics
// A-tile (bf16, regular 2D) is moved by the Tensor Data Mover when available:
//   tile 32 x 128 rows, data_size=2B, LDS padding 4 DW per 16 DW (80B row
//   stride == ldsA layout), wave 0 issues, S_WAIT_TENSORCNT + barrier publishes.
// grid: (D/BN, CAP/BM, E), block: 256
// ---------------------------------------------------------------------------
__global__ __launch_bounds__(256) void moe_ffn2(
    const bf16* __restrict__ hdn, const float* __restrict__ w2,
    const int* __restrict__ counts, const int* __restrict__ offsets,
    const int* __restrict__ token_ids, const float* __restrict__ token_wts,
    float* __restrict__ out)
{
    const int e    = blockIdx.z;
    const int cnt  = counts[e];
    const int tile = blockIdx.y;
    if (tile * BM >= cnt) return;
    const int d0  = blockIdx.x * BN;
    const int off = offsets[e];
    const float* W2 = w2 + (size_t)e * H_DIM * D_DIM;

    __shared__ __align__(16) uint32_t As[BM * 20];
    __shared__ __align__(16) uint32_t Bs[BN * 20];
    __shared__ int   sTok[BM];
    __shared__ float sWt[BM];

    const int tid  = threadIdx.x;
    const int lane = tid & 31;
    const int wave = tid >> 5;
    const int wr   = wave & 3;
    const int wc   = wave >> 2;

    if (tid < BM) {
        int gr = tile * BM + tid;
        if (gr < cnt) { sTok[tid] = token_ids[e * CAP + gr]; sWt[tid] = token_wts[e * CAP + gr]; }
        else          { sTok[tid] = 0;                       sWt[tid] = 0.f; }
    }

#if MOE_TDM
    // ---- Tensor DMA descriptor (uniform, per ISA cdna5 §8) ----
    const uint64_t hdn_tile_base =
        (uint64_t)(uintptr_t)(hdn + (size_t)(off + tile * BM) * H_DIM);
    const uint32_t lds_a_addr = (uint32_t)(uintptr_t)(&As[0]);   // low 32b = LDS byte addr
    i32x8 g1;
    g1[0] = (1 << 16)            // data_size = 2 bytes
          | (1 << 20)            // pad_enable
          | (3 << 22)            // pad_interval: 16 DWORDs (one 64B tile row)
          | (3 << 25);           // pad_amount: 4 DWORDs (16B) -> 80B LDS row stride
    g1[1] = (H_DIM & 0xFFFF) << 16;          // tensor_dim0[15:0]
    g1[2] = (H_DIM >> 16) | ((HDN_ROWS & 0xFFFF) << 16);  // tensor_dim0 hi | tensor_dim1 lo
    g1[3] = (HDN_ROWS >> 16) | (BK << 16);   // tensor_dim1 hi | tile_dim0 = 32
    g1[4] = BM;                              // tile_dim1 = 128 rows (tile_dim2 = 0)
    g1[5] = H_DIM;                           // tensor_dim0_stride lo
    g1[6] = 0;                               // stride0 hi | stride1 lo
    g1[7] = 0;                               // stride1 hi
    i32x4 g2; g2[0] = 0; g2[1] = 0; g2[2] = 0; g2[3] = 0;   // 2D tensor: unused
    i32x4 g3 = g2;
    i32x8 g4;                                                // unused trailing group (this
    g4[0] = 0; g4[1] = 0; g4[2] = 0; g4[3] = 0;              // toolchain's 6-arg builtin)
    g4[4] = 0; g4[5] = 0; g4[6] = 0; g4[7] = 0;
#else
    const int arow  = tid >> 1;
    const int ahalf = tid & 1;
    const bf16* arp = hdn + (size_t)(off + tile * BM + arow) * H_DIM + ahalf * 16;
#endif

    const int bc = tid & 63;
    const int bg = tid >> 6;

    v8f z8 = {0.f,0.f,0.f,0.f,0.f,0.f,0.f,0.f};
    v8f c[2][2];
#pragma unroll
    for (int i = 0; i < 2; ++i) { c[i][0] = z8; c[i][1] = z8; }

    for (int k0 = 0; k0 < H_DIM; k0 += BK) {
#if MOE_TDM
        if (wave == 0) {
            uint64_t ga = hdn_tile_base + (uint64_t)k0 * 2u;  // tile start (bytes)
            u32x4 g0;
            g0[0] = 1u;                                       // count = 1 valid descriptor
            g0[1] = lds_a_addr;                               // LDS destination
            g0[2] = (uint32_t)ga;                             // global_addr[31:0]
            g0[3] = ((uint32_t)(ga >> 32) & 0x01FFFFFFu)      // global_addr[56:32]
                  | 0x80000000u;                              // type = 2 ("image")
            __builtin_amdgcn_tensor_load_to_lds(g0, g1, g2, g3, g4, 0);
        }
#else
        {
            const v4u* ap = (const v4u*)(arp + k0);           // hdn already bf16
            v4u a0 = ap[0];
            v4u a1 = ap[1];
            *(v4u*)(As + arow * 20 + ahalf * 8)     = a0;
            *(v4u*)(As + arow * 20 + ahalf * 8 + 4) = a1;
        }
#endif
        float bv[8];
        const size_t gbase = (size_t)(k0 + bg * 8) * D_DIM + d0 + bc;
#pragma unroll
        for (int i = 0; i < 8; ++i) bv[i] = W2[gbase + (size_t)i * D_DIM];
        if (k0 + BK < H_DIM)
            __builtin_prefetch(&W2[gbase + (size_t)BK * D_DIM], 0, 1);
        v4u bp;
        bp.x = pk2bf(bv[0],bv[1]); bp.y = pk2bf(bv[2],bv[3]);
        bp.z = pk2bf(bv[4],bv[5]); bp.w = pk2bf(bv[6],bv[7]);
        *(v4u*)(Bs + bc * 20 + bg * 4) = bp;
#if MOE_TDM
        if (wave == 0) __builtin_amdgcn_s_wait_tensorcnt(0);  // TDM done -> LDS valid
#endif
        __syncthreads();

        v16bf fa0 = ldsA(As, wr * 32 + 0,  lane);
        v16bf fa1 = ldsA(As, wr * 32 + 16, lane);
#pragma unroll
        for (int nt = 0; nt < 2; ++nt) {
            v16bf fb = ldsB(Bs, wc * 32 + nt * 16, lane);
            c[0][nt] = WMMA_BF16(fa0, fb, c[0][nt]);
            c[1][nt] = WMMA_BF16(fa1, fb, c[1][nt]);
        }
        __syncthreads();
    }

    // ---- epilogue: weighted scatter-add (each token hit by 2 experts) ----
    const int colb = d0 + wc * 32 + (lane & 15);
    const int rsub = (lane >> 4) * 8;
#pragma unroll
    for (int mt = 0; mt < 2; ++mt) {
        const int lrow = wr * 32 + mt * 16 + rsub;
#pragma unroll
        for (int nt = 0; nt < 2; ++nt) {
            v8f acc = c[mt][nt];
            const int col = colb + nt * 16;
#pragma unroll
            for (int i = 0; i < 8; ++i) {
                int lr = lrow + i;
                int gr = tile * BM + lr;
                if (gr < cnt)
                    atomicAdd(out + (size_t)sTok[lr] * D_DIM + col, acc[i] * sWt[lr]);
            }
        }
    }
}

// ---------------------------------------------------------------------------
extern "C" void kernel_launch(void* const* d_in, const int* in_sizes, int n_in,
                              void* d_out, int out_size, void* d_ws, size_t ws_size,
                              hipStream_t stream)
{
    const float* x  = (const float*)d_in[0];
    const float* wg = (const float*)d_in[1];
    const float* bg = (const float*)d_in[2];
    const float* w1 = (const float*)d_in[3];
    const float* w2 = (const float*)d_in[4];
    const float* w3 = (const float*)d_in[5];
    float* out = (float*)d_out;

    uint8_t* ws = (uint8_t*)d_ws;
    int*   counts    = (int*)ws;                                    //  8 ints
    int*   offsets   = (int*)(ws + 64);                             //  8 ints
    int*   token_ids = (int*)(ws + 256);                            //  E*CAP ints (128 KB)
    float* token_wts = (float*)(ws + 256 + (size_t)E_NUM*CAP*4);    //  E*CAP f32  (128 KB)
    bf16*  hdn       = (bf16*)(ws + 256 + (size_t)E_NUM*CAP*8);     //  HDN_ROWS*H bf16 (~94 MB)

    moe_zero<<<2048, 256, 0, stream>>>(out, counts);
    moe_gate<<<NT / 8, 256, 0, stream>>>(x, wg, bg, counts, token_ids, token_wts);
    moe_offsets<<<1, 32, 0, stream>>>(counts, offsets);

    dim3 g1(H_DIM / BN, CAP / BM, E_NUM);
    moe_ffn1<<<g1, 256, 0, stream>>>(x, w1, w3, counts, offsets, token_ids, hdn);

    dim3 g2(D_DIM / BN, CAP / BM, E_NUM);
    moe_ffn2<<<g2, 256, 0, stream>>>(hdn, w2, counts, offsets, token_ids, token_wts, out);
}
